// MotionDecoder_55173149885017
// MI455X (gfx1250) — matrix operand order repaired and checked by reading, not compiled, not run on hardware
//
#include <hip/hip_runtime.h>
#include <hip/hip_bf16.h>
#include <stdint.h>

// ---------------------------------------------------------------------------
// MotionDecoder LSTM for MI455X (gfx1250).
//  * bf16 WMMA (v_wmma_f32_16x16x32_bf16) GEMM core, f32 accumulation/state.
//  * Persistent kernel: 16 WGs x 256 threads (wave32 -> 8 waves/WG), each WG
//    owns 32 hidden columns (all four gates). 2 grid barriers per frame.
//  * Weights converted once per launch to bf16 in WMMA B-fragment order.
//  * pose feedback folded into layer-0 weights (W_pe = W_ih0_pose @ dec_W0),
//    so decoder heads run off the critical path.
// ---------------------------------------------------------------------------

typedef __bf16 bf16;
typedef __attribute__((ext_vector_type(16))) __bf16 v16bf;
typedef __attribute__((ext_vector_type(8)))  __bf16 v8bf;
typedef __attribute__((ext_vector_type(8)))  float  v8f;

#define DEVINL __device__ __forceinline__

constexpr int B_  = 64;     // batch
constexpr int F_  = 512;    // frames
constexpr int D_  = 512;    // hidden
constexpr int KT0 = 48;     // layer0 fused K = 1536 (= x512 | h0 512 | h1 512)
constexpr int KT1 = 32;     // layer1 K = 1024 (= h0new | h1prev)
constexpr int KTD = 16;     // decoder K = 512
constexpr int NTD = 11;     // decoder n-tiles (176 cols, 161 real)
constexpr int NWG = 16;
constexpr int TPB = 256;
constexpr int GPITCH = 34;  // gate LDS pitch (dwords) -> bank-conflict free

// output segment offsets (pp, ps, pc, pct concatenated flat)
constexpr size_t O0 = 0;
constexpr size_t O1 = (size_t)B_*F_*144;
constexpr size_t O2 = O1 + (size_t)B_*F_*10;
constexpr size_t O3 = O2 + (size_t)B_*F_*3;

// ---- workspace layout (bytes, all 256-aligned) ----
constexpr size_t OFF_BAR   = 0;                                    // 256 B barrier
constexpr size_t OFF_WSW0  = 256;                                  // L0 swizzled bf16
constexpr size_t SZ_WSW0   = (size_t)128*KT0*512*2;
constexpr size_t OFF_WSW1  = OFF_WSW0 + SZ_WSW0;                   // L1 swizzled bf16
constexpr size_t SZ_WSW1   = (size_t)128*KT1*512*2;
constexpr size_t OFF_WSWD  = OFF_WSW1 + SZ_WSW1;                   // dec swizzled bf16
constexpr size_t SZ_WSWD   = (size_t)NTD*KTD*512*2;
constexpr size_t OFF_BIAS0 = OFF_WSWD + SZ_WSWD;                   // 2048 f32
constexpr size_t OFF_BIAS1 = OFF_BIAS0 + 2048*4;
constexpr size_t OFF_BIASD = OFF_BIAS1 + 2048*4;                   // 256 f32 (176 used)
constexpr size_t OFF_BPE   = OFF_BIASD + 256*4;                    // 2048 f32
constexpr size_t OFF_WPE   = OFF_BPE + 2048*4;                     // 2048x512 f32
constexpr size_t SZ_WPE    = (size_t)2048*512*4;
constexpr size_t OFF_H0A   = OFF_WPE + SZ_WPE;                     // h ping-pong, bf16
constexpr size_t OFF_H0B   = OFF_H0A + (size_t)B_*D_*2;
constexpr size_t OFF_H1A   = OFF_H0B + (size_t)B_*D_*2;
constexpr size_t OFF_H1B   = OFF_H1A + (size_t)B_*D_*2;
constexpr size_t OFF_C0    = OFF_H1B + (size_t)B_*D_*2;            // f32 cell state
constexpr size_t OFF_C1    = OFF_C0 + (size_t)B_*D_*4;
constexpr size_t OFF_TMP1  = OFF_C1 + (size_t)B_*D_*4;             // MLP temps
constexpr size_t OFF_TMP2  = OFF_TMP1 + (size_t)B_*D_*4;
constexpr size_t OFF_G0X   = OFF_TMP2 + (size_t)B_*1024*4;         // t=0 correction
constexpr size_t WS_NEED   = OFF_G0X + (size_t)B_*2048*4;          // ~16.4 MB

// LDS: fragment-major activations + gate buffer
constexpr size_t ACT_BYTES  = (size_t)4*KT0*32*16*2;               // 196608
constexpr size_t SMEM_TOTAL = ACT_BYTES + (size_t)4*64*GPITCH*4;   // 231424 < 320KB

__constant__ int MJ[12] = {1, 2, 4, 5, 7, 8, 16, 17, 18, 19, 20, 21};

// ---------------------------------------------------------------------------
// grid-wide barrier (sense via generation counter in ws; zeroed per launch)
DEVINL void grid_barrier(unsigned* bar)
{
    __threadfence();
    __syncthreads();
    if (threadIdx.x == 0) {
        unsigned gen = __hip_atomic_load(&bar[1], __ATOMIC_RELAXED, __HIP_MEMORY_SCOPE_AGENT);
        unsigned arr = __hip_atomic_fetch_add(&bar[0], 1u, __ATOMIC_ACQ_REL, __HIP_MEMORY_SCOPE_AGENT);
        if (arr == NWG - 1) {
            __hip_atomic_store(&bar[0], 0u, __ATOMIC_RELAXED, __HIP_MEMORY_SCOPE_AGENT);
            __hip_atomic_fetch_add(&bar[1], 1u, __ATOMIC_RELEASE, __HIP_MEMORY_SCOPE_AGENT);
        } else {
            while (__hip_atomic_load(&bar[1], __ATOMIC_ACQUIRE, __HIP_MEMORY_SCOPE_AGENT) == gen)
                __builtin_amdgcn_s_sleep(2);
        }
    }
    __syncthreads();
    __threadfence();
}

// ---------------------------------------------------------------------------
// Fill LDS activations in WMMA A-fragment order.
//   mode 0 (layer0): K regions [0,512)=x_t fp32->bf16, [512,1024)=hA, [1024,1536)=hB
//   mode 1 (layer1): [0,512)=hA (h0 new), [512,1024)=hB (h1 prev);  dstKT=32
//   mode 2 (final decode): only k-tiles 32..47 from hB (h1), phase-A layout
DEVINL void fill_act(bf16* __restrict__ act, int mode,
                     const float* __restrict__ x, int t,
                     const bf16* __restrict__ hA, const bf16* __restrict__ hB, int tid)
{
    const int KTf   = (mode == 2) ? 16 : (mode == 1 ? 32 : 48);
    const int dstKT = (mode == 1) ? 32 : 48;
    const int ktOff = (mode == 2) ? 32 : 0;
    const int total = 4 * KTf * 32;
    for (int s = tid; s < total; s += TPB) {
        const int lane_ = s & 31;
        const int kt    = ((s >> 5) % KTf) + ktOff;
        const int mt    = s / (32 * KTf);
        const int m     = mt*16 + (lane_ & 15);
        const int hh    = lane_ >> 4;
        bf16* dst = act + ((size_t)(mt*dstKT + kt)*32 + lane_)*16;
#pragma unroll
        for (int run = 0; run < 2; ++run) {
            const int kb = kt*32 + run*16 + 8*hh;   // 8 consecutive K per run
            v8bf v;
            if (mode == 0) {
                if (kb < 512) {
                    v8f xv = *(const v8f*)(x + ((size_t)m*F_ + t)*D_ + kb);
#pragma unroll
                    for (int i = 0; i < 8; ++i) v[i] = (bf16)xv[i];
                } else if (kb < 1024) {
                    v = *(const v8bf*)(hA + (size_t)m*D_ + (kb - 512));
                } else {
                    v = *(const v8bf*)(hB + (size_t)m*D_ + (kb - 1024));
                }
            } else if (mode == 1) {
                if (kb < 512) v = *(const v8bf*)(hA + (size_t)m*D_ + kb);
                else          v = *(const v8bf*)(hB + (size_t)m*D_ + (kb - 512));
            } else {
                v = *(const v8bf*)(hB + (size_t)m*D_ + (kb - 1024));
            }
            *(v8bf*)(dst + run*8) = v;
        }
    }
}

// ---------------------------------------------------------------------------
// One gate GEMM: this wave computes a 64x16 slice (4 M-tiles) of the 64x2048
// pre-activation, accumulating over KT k-tiles, then deposits (+bias) in LDS.
template <int KT>
DEVINL void gemm_gates(const bf16* __restrict__ act, const bf16* __restrict__ Wsw,
                       const float* __restrict__ bias, float* __restrict__ gbuf,
                       int wg, int wave, int lane)
{
    const int gate = wave >> 1, sub = wave & 1;
    const int gn   = gate*32 + wg*2 + sub;          // global n-tile 0..127
    const v16bf* Bp = (const v16bf*)Wsw + (size_t)gn*(KT*32);
    const v16bf* Ap = (const v16bf*)act;
    v8f acc[4];
#pragma unroll
    for (int mt = 0; mt < 4; ++mt) {
#pragma unroll
        for (int i = 0; i < 8; ++i) acc[mt][i] = 0.f;
    }
    for (int k = 0; k < KT; ++k) {
        v16bf b = Bp[(size_t)k*32 + lane];          // 2x global_load_b128 (L2-resident)
#pragma unroll
        for (int mt = 0; mt < 4; ++mt) {
            v16bf a = Ap[(size_t)(mt*KT + k)*32 + lane];  // 2x ds_load_b128
            acc[mt] = __builtin_amdgcn_wmma_f32_16x16x32_bf16(
                false, a, false, b, (short)0, acc[mt], false, false);
        }
    }
    const int nloc = lane & 15, half = lane >> 4;
    const int colw = sub*16 + nloc;                 // col within WG's 32 columns
    const float bv = bias[gate*512 + wg*32 + colw];
#pragma unroll
    for (int mt = 0; mt < 4; ++mt) {
#pragma unroll
        for (int r = 0; r < 8; ++r) {
            const int m = mt*16 + half*8 + r;       // C/D layout: lane->N, vgpr->M
            gbuf[((gate*64) + m)*GPITCH + colw] = acc[mt][r] + bv;
        }
    }
}

// ---------------------------------------------------------------------------
// LSTM pointwise: c' = sig(f)*c + sig(i)*tanh(g); h' = sig(o)*tanh(c')
DEVINL void combine_gates(const float* __restrict__ gbuf, float* __restrict__ cst,
                          bf16* __restrict__ hout, const float* __restrict__ g0x,
                          int wg, int tid)
{
    const int colw = tid & 31;
    const int m0   = tid >> 5;
    const int cg   = wg*32 + colw;                  // hidden column 0..511
#pragma unroll
    for (int q = 0; q < 8; ++q) {
        const int m = q*8 + m0;
        float gi = gbuf[(0*64 + m)*GPITCH + colw];
        float gf = gbuf[(1*64 + m)*GPITCH + colw];
        float gg = gbuf[(2*64 + m)*GPITCH + colw];
        float go = gbuf[(3*64 + m)*GPITCH + colw];
        if (g0x) {   // t==0: replace folded pose term with true init-pose term
            const float* gx = g0x + (size_t)m*2048;
            gi += gx[cg]; gf += gx[512 + cg]; gg += gx[1024 + cg]; go += gx[1536 + cg];
        }
        const float i_ = 1.f/(1.f + __expf(-gi));
        const float f_ = 1.f/(1.f + __expf(-gf));
        const float g_ = tanhf(gg);
        const float o_ = 1.f/(1.f + __expf(-go));
        const float c  = f_*cst[(size_t)m*D_ + cg] + i_*g_;
        cst[(size_t)m*D_ + cg] = c;
        hout[(size_t)m*D_ + cg] = (bf16)(o_*tanhf(c));
    }
}

// ---------------------------------------------------------------------------
// Decoder heads for frame t, consuming h1 tiles already staged in LDS
// (phase-A layout k-tiles 32..47). 44 C-tiles spread over the 16 WGs.
DEVINL void decode_frame(int t, const bf16* __restrict__ act,
                         const bf16* __restrict__ Wswd, const float* __restrict__ biasd,
                         float* __restrict__ out, int wg, int wave, int lane)
{
    for (int tIdx = wg; tIdx < 4*NTD; tIdx += NWG) {
        if ((tIdx >> 4) != wave) continue;          // wave-uniform branch
        const int mt = tIdx / NTD;
        const int nd = tIdx % NTD;
        const v16bf* Bp = (const v16bf*)Wswd + (size_t)nd*(KTD*32);
        const v16bf* Ap = (const v16bf*)act;
        v8f acc;
#pragma unroll
        for (int i = 0; i < 8; ++i) acc[i] = 0.f;
        for (int k = 0; k < KTD; ++k) {
            v16bf b = Bp[(size_t)k*32 + lane];
            v16bf a = Ap[(size_t)(mt*KT0 + 32 + k)*32 + lane];
            acc = __builtin_amdgcn_wmma_f32_16x16x32_bf16(
                false, a, false, b, (short)0, acc, false, false);
        }
        const int c    = nd*16 + (lane & 15);
        const int half = lane >> 4;
        const float bv = biasd[c];
#pragma unroll
        for (int r = 0; r < 8; ++r) {
            const int m = mt*16 + half*8 + r;
            const float v = acc[r] + bv;
            const size_t bt = (size_t)m*F_ + t;
            if      (c < 144) out[O0 + bt*144 + c]         = v;
            else if (c < 154) out[O1 + bt*10  + (c - 144)] = v;
            else if (c < 157) out[O2 + bt*3   + (c - 154)] = v;
            else if (c < 161) out[O3 + bt*4   + (c - 157)] = v;
        }
    }
}

// ---------------------------------------------------------------------------
__global__ __launch_bounds__(TPB, 1)
void lstm_persistent(const float* __restrict__ x, float* __restrict__ out,
                     uint8_t* __restrict__ ws)
{
    extern __shared__ char smem[];
    bf16*  act  = (bf16*)smem;
    float* gbuf = (float*)(smem + ACT_BYTES);

    const bf16*  Wsw0  = (const bf16*)(ws + OFF_WSW0);
    const bf16*  Wsw1  = (const bf16*)(ws + OFF_WSW1);
    const bf16*  Wswd  = (const bf16*)(ws + OFF_WSWD);
    const float* bias0 = (const float*)(ws + OFF_BIAS0);
    const float* bias1 = (const float*)(ws + OFF_BIAS1);
    const float* biasd = (const float*)(ws + OFF_BIASD);
    const float* g0x   = (const float*)(ws + OFF_G0X);
    float* c0st = (float*)(ws + OFF_C0);
    float* c1st = (float*)(ws + OFF_C1);
    bf16*  h0s[2] = { (bf16*)(ws + OFF_H0A), (bf16*)(ws + OFF_H0B) };
    bf16*  h1s[2] = { (bf16*)(ws + OFF_H1A), (bf16*)(ws + OFF_H1B) };
    unsigned* bar = (unsigned*)(ws + OFF_BAR);

    const int tid  = threadIdx.x;
    const int wg   = blockIdx.x;
    const int wave = tid >> 5;
    const int lane = tid & 31;

    for (int t = 0; t < F_; ++t) {
        const int p = t & 1;
        // ---- layer 0 (+ decoder of frame t-1 off the critical path) ----
        fill_act(act, 0, x, t, h0s[p], h1s[p], tid);
        __syncthreads();
        gemm_gates<KT0>(act, Wsw0, bias0, gbuf, wg, wave, lane);
        if (t > 0) decode_frame(t - 1, act, Wswd, biasd, out, wg, wave, lane);
        __syncthreads();
        combine_gates(gbuf, c0st, h0s[1 - p], (t == 0) ? g0x : nullptr, wg, tid);
        grid_barrier(bar);
        // ---- layer 1 ----
        fill_act(act, 1, nullptr, t, h0s[1 - p], h1s[p], tid);
        __syncthreads();
        gemm_gates<KT1>(act, Wsw1, bias1, gbuf, wg, wave, lane);
        __syncthreads();
        combine_gates(gbuf, c1st, h1s[1 - p], nullptr, wg, tid);
        grid_barrier(bar);
    }
    // ---- decoder for the last frame ----
    fill_act(act, 2, nullptr, 0, nullptr, h1s[0], tid);
    __syncthreads();
    decode_frame(F_ - 1, act, Wswd, biasd, out, wg, wave, lane);
}

// ===========================================================================
// One-time (per launch) preparation kernels: init MLP, weight fold/convert.
// ===========================================================================
__global__ void k_mlp1(const float* __restrict__ initp, const float* __restrict__ W,
                       const float* __restrict__ b, float* __restrict__ o)
{
    for (int idx = blockIdx.x*blockDim.x + threadIdx.x; idx < 64*512;
         idx += gridDim.x*blockDim.x) {
        const int m = idx >> 9, oo = idx & 511;
        const float* wr = W + oo*72;
        const float* im = initp + m*144;
        float s = b[oo];
#pragma unroll
        for (int j = 0; j < 12; ++j) {
            const int base = MJ[j]*6;
#pragma unroll
            for (int d = 0; d < 6; ++d) s += im[base + d] * wr[j*6 + d];
        }
        o[idx] = fmaxf(s, 0.f);
    }
}

__global__ void k_mlp2(const float* __restrict__ t1, const float* __restrict__ W,
                       const float* __restrict__ b, float* __restrict__ o)
{
    for (int idx = blockIdx.x*blockDim.x + threadIdx.x; idx < 64*1024;
         idx += gridDim.x*blockDim.x) {
        const int m = idx >> 10, oo = idx & 1023;
        const float* wr = W + (size_t)oo*512;
        const float* im = t1 + m*512;
        float s = b[oo];
        for (int k = 0; k < 512; ++k) s += im[k] * wr[k];
        o[idx] = fmaxf(s, 0.f);
    }
}

__global__ void k_mlp3(const float* __restrict__ t2, const float* __restrict__ W,
                       const float* __restrict__ b, bf16* h0, bf16* h1,
                       float* c0, float* c1)
{
    for (int idx = blockIdx.x*blockDim.x + threadIdx.x; idx < 64*2048;
         idx += gridDim.x*blockDim.x) {
        const int m = idx >> 11, oo = idx & 2047;
        const float* wr = W + (size_t)oo*1024;
        const float* im = t2 + m*1024;
        float s = b[oo];
        for (int k = 0; k < 1024; ++k) s += im[k] * wr[k];
        // hc.reshape(b,2,L,D): [0:512)=h0, [512:1024)=h1, [1024:1536)=c0, rest c1
        if      (oo <  512) h0[(size_t)m*512 + oo]          = (bf16)s;
        else if (oo < 1024) h1[(size_t)m*512 + (oo - 512)]  = (bf16)s;
        else if (oo < 1536) c0[(size_t)m*512 + (oo - 1024)] = s;
        else                c1[(size_t)m*512 + (oo - 1536)] = s;
    }
}

__global__ void k_wpe(const float* __restrict__ Wih0, const float* __restrict__ dW0,
                      float* __restrict__ Wpe)
{
    for (int idx = blockIdx.x*blockDim.x + threadIdx.x; idx < 2048*512;
         idx += gridDim.x*blockDim.x) {
        const int n = idx >> 9, c = idx & 511;
        const float* wr = Wih0 + (size_t)n*656 + 512;
        float s = 0.f;
        for (int j = 0; j < 144; ++j) s += wr[j] * dW0[(size_t)j*512 + c];
        Wpe[idx] = s;
    }
}

__global__ void k_bias(const float* Wih0, const float* db0,
                       const float* bih0, const float* bhh0,
                       const float* bih1, const float* bhh1,
                       const float* db1, const float* db2, const float* db3,
                       float* bias0, float* bias1, float* biasd, float* bpe)
{
    const int idx = blockIdx.x*blockDim.x + threadIdx.x;
    if (idx < 2048) {
        const float* wr = Wih0 + (size_t)idx*656 + 512;
        float pe = 0.f;
        for (int j = 0; j < 144; ++j) pe += wr[j] * db0[j];
        bpe[idx]   = pe;
        bias0[idx] = bih0[idx] + bhh0[idx] + pe;   // pose bias folded in
        bias1[idx] = bih1[idx] + bhh1[idx];
    }
    if (idx < 176) {
        float v = 0.f;
        if      (idx < 144) v = db0[idx];
        else if (idx < 154) v = db1[idx - 144];
        else if (idx < 157) v = db2[idx - 154];
        else if (idx < 161) v = db3[idx - 157];
        biasd[idx] = v;
    }
}

// Convert + swizzle weights into WMMA B-fragment order:
// element e = ((gn*KT + kt)*32 + lane)*16 + i ; n = gn*16+(lane&15) ;
// k = kt*32 + (i&7) + 8*(lane>>4) + 16*(i>>3)   (B layout: table order, matches A)
__global__ void k_swz(int mode, bf16* __restrict__ dst,
                      const float* s0, const float* s1, const float* s2, const float* s3,
                      int NTl, int KTl)
{
    const long total = (long)NTl * KTl * 512;
    for (long e = blockIdx.x*(long)blockDim.x + threadIdx.x; e < total;
         e += (long)gridDim.x*blockDim.x) {
        const int  i    = (int)(e & 15);
        const int  lane = (int)((e >> 4) & 31);
        const long kt   = (e >> 9) % KTl;
        const long gn   = e / ((long)KTl * 512);
        const int  n    = (int)gn*16 + (lane & 15);
        const int  k    = (int)kt*32 + (i & 7) + 8*(lane >> 4) + 16*(i >> 3);
        float v = 0.f;
        if (mode == 0) {          // [W_ih0_x | W_hh0 | W_pe], K=1536
            if      (k <  512) v = s0[(long)n*656 + k];
            else if (k < 1024) v = s1[(long)n*512 + (k - 512)];
            else               v = s2[(long)n*512 + (k - 1024)];
        } else if (mode == 1) {   // [W_ih1 | W_hh1], K=1024
            v = (k < 512) ? s0[(long)n*512 + k] : s1[(long)n*512 + (k - 512)];
        } else {                  // decoder heads stacked, rows >=161 zero
            if      (n < 144) v = s0[(long)n*512 + k];
            else if (n < 154) v = s1[(long)(n - 144)*512 + k];
            else if (n < 157) v = s2[(long)(n - 154)*512 + k];
            else if (n < 161) v = s3[(long)(n - 157)*512 + k];
        }
        dst[e] = (bf16)v;
    }
}

// t=0 gate correction: + pose0@W_pᵀ  - h1init@W_peᵀ - b_pe
__global__ void k_g0x(const float* __restrict__ Wih0, const float* __restrict__ initp,
                      const float* __restrict__ Wpe, const float* __restrict__ bpe,
                      const bf16* __restrict__ h1, float* __restrict__ g0x)
{
    for (int idx = blockIdx.x*blockDim.x + threadIdx.x; idx < 64*2048;
         idx += gridDim.x*blockDim.x) {
        const int m = idx >> 11, n = idx & 2047;
        float s = -bpe[n];
        const float* wr = Wih0 + (size_t)n*656 + 512;
        const float* im = initp + m*144;
        for (int j = 0; j < 144; ++j) s += wr[j] * im[j];
        const float* wp = Wpe + (size_t)n*512;
        const bf16*  hm = h1 + (size_t)m*512;
        for (int c = 0; c < 512; ++c) s -= wp[c] * (float)hm[c];
        g0x[idx] = s;
    }
}

// ===========================================================================
extern "C" void kernel_launch(void* const* d_in, const int* in_sizes, int n_in,
                              void* d_out, int out_size, void* d_ws, size_t ws_size,
                              hipStream_t stream)
{
    if (ws_size < WS_NEED) return;   // need ~16.4 MB scratch

    const float* x     = (const float*)d_in[0];
    const float* initp = (const float*)d_in[1];
    const float* niW1  = (const float*)d_in[2];
    const float* nib1  = (const float*)d_in[3];
    const float* niW2  = (const float*)d_in[4];
    const float* nib2  = (const float*)d_in[5];
    const float* niW3  = (const float*)d_in[6];
    const float* nib3  = (const float*)d_in[7];
    const float* Wih0  = (const float*)d_in[8];
    const float* bih0  = (const float*)d_in[9];
    const float* Whh0  = (const float*)d_in[10];
    const float* bhh0  = (const float*)d_in[11];
    const float* Wih1  = (const float*)d_in[12];
    const float* bih1  = (const float*)d_in[13];
    const float* Whh1  = (const float*)d_in[14];
    const float* bhh1  = (const float*)d_in[15];
    const float* dW0   = (const float*)d_in[16];
    const float* db0   = (const float*)d_in[17];
    const float* dW1   = (const float*)d_in[18];
    const float* db1   = (const float*)d_in[19];
    const float* dW2   = (const float*)d_in[20];
    const float* db2   = (const float*)d_in[21];
    const float* dW3   = (const float*)d_in[22];
    const float* db3   = (const float*)d_in[23];
    float*   out = (float*)d_out;
    uint8_t* ws  = (uint8_t*)d_ws;

    // reset grid barrier (state must be clean for every graph replay)
    hipMemsetAsync(ws + OFF_BAR, 0, 256, stream);

    float* tmp1 = (float*)(ws + OFF_TMP1);
    float* tmp2 = (float*)(ws + OFF_TMP2);
    bf16*  h0A  = (bf16*)(ws + OFF_H0A);
    bf16*  h1A  = (bf16*)(ws + OFF_H1A);
    float* c0   = (float*)(ws + OFF_C0);
    float* c1   = (float*)(ws + OFF_C1);
    float* Wpe  = (float*)(ws + OFF_WPE);

    // neural initialization MLP -> (h0, c0, h1, c1)
    k_mlp1<<<128, TPB, 0, stream>>>(initp, niW1, nib1, tmp1);
    k_mlp2<<<256, TPB, 0, stream>>>(tmp1, niW2, nib2, tmp2);
    k_mlp3<<<512, TPB, 0, stream>>>(tmp2, niW3, nib3, h0A, h1A, c0, c1);

    // fold pose feedback: W_pe = W_ih0_pose @ dec_W0 ; biases
    k_wpe<<<4096, TPB, 0, stream>>>(Wih0, dW0, Wpe);
    k_bias<<<8, TPB, 0, stream>>>(Wih0, db0, bih0, bhh0, bih1, bhh1, db1, db2, db3,
                                  (float*)(ws + OFF_BIAS0), (float*)(ws + OFF_BIAS1),
                                  (float*)(ws + OFF_BIASD), (float*)(ws + OFF_BPE));

    // bf16 conversion + B-fragment swizzle
    k_swz<<<4096, TPB, 0, stream>>>(0, (bf16*)(ws + OFF_WSW0), Wih0, Whh0, Wpe, nullptr, 128, KT0);
    k_swz<<<4096, TPB, 0, stream>>>(1, (bf16*)(ws + OFF_WSW1), Wih1, Whh1, nullptr, nullptr, 128, KT1);
    k_swz<<<352,  TPB, 0, stream>>>(2, (bf16*)(ws + OFF_WSWD), dW0, dW1, dW2, dW3, NTD, KTD);

    // t=0 correction term
    k_g0x<<<512, TPB, 0, stream>>>(Wih0, initp, Wpe, (float*)(ws + OFF_BPE), h1A,
                                   (float*)(ws + OFF_G0X));

    // persistent recurrent kernel: 16 WGs, 231 KB LDS each (1 WG per WGP)
    (void)hipFuncSetAttribute((const void*)lstm_persistent,
                              hipFuncAttributeMaxDynamicSharedMemorySize,
                              (int)SMEM_TOTAL);
    lstm_persistent<<<dim3(NWG), dim3(TPB), SMEM_TOTAL, stream>>>(x, out, ws);
}
